// MGKAP_66211215835417
// MI455X (gfx1250) — matrix operand order repaired and checked by reading, not compiled, not run on hardware
//
#include <hip/hip_runtime.h>
#include <hip/hip_bf16.h>
#include <cstdint>
#include <cstddef>

// ---- problem constants (match reference) ----
#define Bc 4096
#define Nn 64
#define Dd 128
#define Hh 4
#define Ll 2
#define SCALE_F 0.17677669529663687f   // DK^-0.5, DK=32

typedef __attribute__((ext_vector_type(16))) __bf16 v16bf;
typedef __attribute__((ext_vector_type(8)))  float  v8f;

#define WMMA_BF16(a, b, c) \
  __builtin_amdgcn_wmma_f32_16x16x32_bf16(false, (a), false, (b), (short)0, (c), false, false)

// Use CDNA5 async global->LDS copies (ASYNCcnt-tracked, no VGPR round-trip)
#define USE_ASYNC_LDS 1

__device__ __forceinline__ void async_copy16(void* lds_dst, const void* gsrc) {
#if USE_ASYNC_LDS
  unsigned l = (unsigned)(uintptr_t)lds_dst;   // low 32 bits of generic LDS ptr = LDS offset
  asm volatile("global_load_async_to_lds_b128 %0, %1, off"
               :: "v"(l), "v"(gsrc) : "memory");
#else
  *(uint4*)lds_dst = *(const uint4*)gsrc;
#endif
}
__device__ __forceinline__ void async_wait_all() {
#if USE_ASYNC_LDS
  asm volatile("s_wait_asynccnt 0x0" ::: "memory");
#endif
}

// ---------------------------------------------------------------------------
// WMMA fragment layouts per CDNA5 ISA 7.12.2 (wave32):
//  A (16x32 bf16): lane<16 -> M=lane, K offs {0..7,16..23}; lane>=16 -> K offs {8..15,24..31}
//  B (32x16 bf16): lane<16 -> N=lane, K offs {0..7,16..23}; lane>=16 -> K offs {8..15,24..31}
//  C (16x16 f32):  VGPR r -> M = r + (lane>=16 ? 8 : 0), N = lane&15
// ---------------------------------------------------------------------------

// A fragment from LDS (row-major 64x128 bf16): two contiguous 16B chunks -> 2x ds_load_b128
__device__ __forceinline__ v16bf load_a_lds(const __bf16* S, int m0, int k0, int lane) {
  int row = m0 + (lane & 15);
  int kb  = k0 + ((lane & 16) ? 8 : 0);
  const __bf16* p = S + row * Dd + kb;
  union { uint4 u[2]; v16bf v; } r;
  r.u[0] = *(const uint4*)(p);
  r.u[1] = *(const uint4*)(p + 16);
  return r.v;
}

// A fragment from global f32 row-major [M x 128] with on-load bf16 conversion
__device__ __forceinline__ v16bf load_a_g(const float* A, int m0, int k0, int lane) {
  int row = m0 + (lane & 15);
  int kb  = k0 + ((lane & 16) ? 8 : 0);
  const float* p = A + (size_t)row * Dd + kb;
  v16bf a;
#pragma unroll
  for (int i = 0; i < 8; ++i) { a[i] = (__bf16)p[i]; a[8 + i] = (__bf16)p[16 + i]; }
  return a;
}

// B fragment from pre-swizzled packed weights: 16 contiguous bf16 per lane (2x b128)
__device__ __forceinline__ v16bf load_w_p(const __bf16* P, int kt, int nt, int lane) {
  return *(const v16bf*)(P + (((size_t)(kt * 8 + nt) * 32) + lane) * 16);
}

// ---------------------------------------------------------------------------
// Pre-swizzle a [ktiles*32 x 128] f32 weight matrix into WMMA B-fragment order.
// grid = ktiles*8 blocks (one per (kt,nt) tile), 512 threads (lane, half-idx).
// ---------------------------------------------------------------------------
__global__ __launch_bounds__(512) void pack_w_kernel(const float* __restrict__ W,
                                                     __bf16* __restrict__ P, int ktiles) {
  int blk  = blockIdx.x;            // kt*8 + nt
  int t    = threadIdx.x;           // 512
  int lane = t & 31, i = t >> 5;    // i = 0..15
  int kt = blk >> 3, nt = blk & 7;
  int row = kt * 32 + ((lane & 16) ? 8 : 0) + (i & 7) + ((i & 8) ? 16 : 0);
  int col = nt * 16 + (lane & 15);
  P[((size_t)blk * 32 + lane) * 16 + i] = (__bf16)W[(size_t)row * Dd + col];
  (void)ktiles;
}

// ---------------------------------------------------------------------------
// prep: f32 -> bf16 mirrors of ent/rel (halves traffic; 128MB fits in 192MB L2)
// ---------------------------------------------------------------------------
__global__ void prep_kernel(const float* __restrict__ ent, const float* __restrict__ rel,
                            __bf16* __restrict__ entb, __bf16* __restrict__ relb, size_t n) {
  size_t i = (size_t)blockIdx.x * blockDim.x + threadIdx.x;
  if (i < n) { entb[i] = (__bf16)ent[i]; relb[i] = (__bf16)rel[i]; }
}

// v0 = mean over N of neighbor_vectors
__global__ void mean_kernel(const float* __restrict__ ent, float* __restrict__ v0) {
  int i = blockIdx.x * blockDim.x + threadIdx.x;      // over B*D
  int b = i >> 7, d = i & 127;
  const float* p = ent + (size_t)b * Nn * Dd + d;
  float s = 0.f;
#pragma unroll 4
  for (int n = 0; n < Nn; ++n) s += p[(size_t)n * Dd];
  v0[i] = s * (1.0f / Nn);
}

// ---------------------------------------------------------------------------
// Generic C[M,128] = (A0@P0 [+ A1@P1]) + bias(es), optional *scale, optional tanh.
// P* are packed bf16 weights (K=128 each). grid = M/16; 8 waves; wave = N-tile.
// ---------------------------------------------------------------------------
__global__ __launch_bounds__(256) void gemm128_kernel(
    const float* __restrict__ A0, const __bf16* __restrict__ P0,
    const float* __restrict__ A1, const __bf16* __restrict__ P1,
    const float* __restrict__ b0, const float* __restrict__ b1,
    float scale, int do_tanh, float* __restrict__ C)
{
  int lane = threadIdx.x & 31;
  int nt   = threadIdx.x >> 5;
  int m0   = blockIdx.x * 16;
  int n0   = nt * 16;
  v8f acc = {};
#pragma unroll
  for (int kt = 0; kt < 4; ++kt)
    acc = WMMA_BF16(load_a_g(A0, m0, kt * 32, lane), load_w_p(P0, kt, nt, lane), acc);
  if (A1 != nullptr) {
#pragma unroll
    for (int kt = 0; kt < 4; ++kt)
      acc = WMMA_BF16(load_a_g(A1, m0, kt * 32, lane), load_w_p(P1, kt, nt, lane), acc);
  }
  int col = n0 + (lane & 15);
  int rb  = m0 + ((lane & 16) ? 8 : 0);
  float bias = (b0 ? b0[col] : 0.0f) + (b1 ? b1[col] : 0.0f);
#pragma unroll
  for (int r = 0; r < 8; ++r) {
    float v = (acc[r] + bias) * scale;
    if (do_tanh) v = tanhf(v);
    C[(size_t)(rb + r) * Dd + col] = v;
  }
}

// ---------------------------------------------------------------------------
// Fused per-b attention layer. One workgroup per b (256 threads = 8 waves).
//  1. async-stage ent/rel rows (64x128 bf16 each) into LDS
//  2. K = [rel|ent]@WK via WMMA (W frags preloaded once); fold Q-dot into scores
//     with ds_add_f32 atomics (K never materialized); + Q.bK; softmax over n/head
//  3. M = sa*ent + sb*rel (bf16, LDS); Vsum = M@WV + (sa+sb)*bV via WMMA
//     (Vsum overlays the dead ent/rel LDS region)
//  4. scrambled reduction: prop[b, c*32+k] = sum_h sum_j w[4j+c,h]*Vsum[4j+c,h*32+k]
// ---------------------------------------------------------------------------
struct SMemAttn {
  union {
    struct { __bf16 sE[Nn * Dd]; __bf16 sR[Nn * Dd]; } in;   // 32 KB
    float sV[Nn * Dd];                                        // 32 KB (reuse)
  } u;
  __bf16 sM[Nn * Dd];      // 16 KB
  float  sSc[Nn * Hh];     // 1 KB
};

__global__ __launch_bounds__(256) void layer_attn_kernel(
    const __bf16* __restrict__ entb, const __bf16* __restrict__ relb,
    const __bf16* __restrict__ WKp,  // packed, 8 kt-tiles (rows 0..127=rel, 128..255=ent)
    const float* __restrict__ bK,
    const __bf16* __restrict__ WVp,  // packed, 4 kt-tiles
    const float* __restrict__ bV,
    const float* __restrict__ alpha, const float* __restrict__ beta,
    const float* __restrict__ Qs,    // [B,128], already *SCALE
    float* __restrict__ prop)        // [B,128]
{
  __shared__ SMemAttn sm;
  int b = blockIdx.x, t = threadIdx.x;
  int lane = t & 31, wave = t >> 5;

  // ---- stage tiles via async global->LDS (16B granularity, per-lane addresses) ----
  {
    const __bf16* ge = entb + (size_t)b * Nn * Dd;
    const __bf16* gr = relb + (size_t)b * Nn * Dd;
    for (int i = t; i < Nn * Dd / 8; i += 256) {
      async_copy16((char*)sm.u.in.sE + (size_t)i * 16, ge + (size_t)i * 8);
      async_copy16((char*)sm.u.in.sR + (size_t)i * 16, gr + (size_t)i * 8);
    }
    sm.sSc[t] = 0.0f;     // 256 entries, one per thread
    async_wait_all();
  }
  __syncthreads();

  const float* qrow = Qs + (size_t)b * Dd;

  // ---- scores: K-GEMM tiles with inline Q-dot (LDS float atomics) ----
  {
    int n0  = wave * 16;
    int col = n0 + (lane & 15);
    float qv = qrow[col];
    int h = col >> 5;
    v16bf wkR[4], wkE[4];                 // hoisted: depend only on (kt, wave)
#pragma unroll
    for (int kt = 0; kt < 4; ++kt) {
      wkR[kt] = load_w_p(WKp, kt,     wave, lane);
      wkE[kt] = load_w_p(WKp, 4 + kt, wave, lane);
    }
    for (int mt = 0; mt < 4; ++mt) {
      v8f acc = {};
#pragma unroll
      for (int kt = 0; kt < 4; ++kt)      // rel part: WK rows 0..127
        acc = WMMA_BF16(load_a_lds(sm.u.in.sR, mt * 16, kt * 32, lane), wkR[kt], acc);
#pragma unroll
      for (int kt = 0; kt < 4; ++kt)      // ent part: WK rows 128..255
        acc = WMMA_BF16(load_a_lds(sm.u.in.sE, mt * 16, kt * 32, lane), wkE[kt], acc);
      int rb = mt * 16 + ((lane & 16) ? 8 : 0);
#pragma unroll
      for (int r = 0; r < 8; ++r)
        atomicAdd(&sm.sSc[(rb + r) * Hh + h], qv * acc[r]);
    }
  }
  __syncthreads();

  // ---- bias term + softmax over n (per head) ----
  {
    int h = t & 3;
    float qb = 0.0f;
#pragma unroll
    for (int k = 0; k < 32; ++k) qb += qrow[h * 32 + k] * bK[h * 32 + k];
    float s = sm.sSc[t] + qb;
    sm.sSc[t] = s;          // own slot
    __syncthreads();
    float mx = -1e30f;
    for (int j = 0; j < Nn; ++j) mx = fmaxf(mx, sm.sSc[j * Hh + h]);
    float den = 0.0f;
    for (int j = 0; j < Nn; ++j) den += __expf(sm.sSc[j * Hh + h] - mx);
    float e = __expf(s - mx);
    __syncthreads();        // all reads done before overwrite
    sm.sSc[t] = e / den;
  }

  // ---- M = sa*ent + sb*rel; Vsum = M@WV + (sa+sb)*bV ----
  float sa = 0.f, sb = 0.f;
#pragma unroll
  for (int i = 0; i < Hh; ++i) { sa += alpha[i]; sb += beta[i]; }
  sa *= (1.0f / Hh); sb *= (1.0f / Hh);
  for (int i = t; i < Nn * Dd; i += 256)
    sm.sM[i] = (__bf16)(sa * (float)sm.u.in.sE[i] + sb * (float)sm.u.in.sR[i]);
  __syncthreads();          // sE/sR dead -> union region becomes sV

  {
    int n0  = wave * 16;
    int col = n0 + (lane & 15);
    float bvs = (sa + sb) * bV[col];
    v16bf wv[4];
#pragma unroll
    for (int kt = 0; kt < 4; ++kt) wv[kt] = load_w_p(WVp, kt, wave, lane);
    for (int mt = 0; mt < 4; ++mt) {
      v8f acc = {};
#pragma unroll
      for (int kt = 0; kt < 4; ++kt)
        acc = WMMA_BF16(load_a_lds(sm.sM, mt * 16, kt * 32, lane), wv[kt], acc);
      int rb = mt * 16 + ((lane & 16) ? 8 : 0);
#pragma unroll
      for (int r = 0; r < 8; ++r)
        sm.u.sV[(size_t)(rb + r) * Dd + col] = acc[r] + bvs;
    }
  }
  __syncthreads();

  // ---- scrambled permute+reshape+sum reduction (derived closed form) ----
  if (t < Dd) {
    int c = t >> 5, k = t & 31;
    float accp = 0.f;
    for (int j = 0; j < 16; ++j) {
      int n = 4 * j + c;
#pragma unroll
      for (int h = 0; h < Hh; ++h)
        accp += sm.sSc[n * Hh + h] * sm.u.sV[n * Dd + h * 32 + k];
    }
    prop[(size_t)b * Dd + t] = accp;
  }
}

// ---------------------------------------------------------------------------
// Hierarchical fusion (VALU; ~1 GFLOP total, not worth WMMA):
// gamma = softmax_i( tanh([u|v_i]@W_layer + b_layer) . w_layer )
// tmp = self + sum_i gamma_i * v_i     (one block per b, 128 threads)
// ---------------------------------------------------------------------------
__global__ __launch_bounds__(128) void fusion_kernel(
    const float* __restrict__ u, const float* __restrict__ self,
    const float* __restrict__ v0, const float* __restrict__ v1, const float* __restrict__ v2,
    const float* __restrict__ Wl, const float* __restrict__ bl, const float* __restrict__ wl,
    float* __restrict__ tmp)
{
  __shared__ float red[Dd];
  __shared__ float g[3];
  int b = blockIdx.x, d = threadIdx.x;
  const float* vs0 = v0 + (size_t)b * Dd;
  const float* vs1 = v1 + (size_t)b * Dd;
  const float* vs2 = v2 + (size_t)b * Dd;
  const float* vsi[3] = { vs0, vs1, vs2 };
  const float* ub = u + (size_t)b * Dd;

  float uacc = bl[d];
  for (int j = 0; j < Dd; ++j) uacc += ub[j] * Wl[(size_t)j * Dd + d];

  for (int i = 0; i < 3; ++i) {
    float s = uacc;
    for (int j = 0; j < Dd; ++j) s += vsi[i][j] * Wl[(size_t)(Dd + j) * Dd + d];
    float hv = tanhf(s);
    red[d] = hv * wl[d];
    __syncthreads();
    for (int off = 64; off > 0; off >>= 1) {
      if (d < off) red[d] += red[d + off];
      __syncthreads();
    }
    if (d == 0) g[i] = red[0];
    __syncthreads();
  }
  float mx = fmaxf(g[0], fmaxf(g[1], g[2]));
  float e0 = __expf(g[0] - mx), e1 = __expf(g[1] - mx), e2 = __expf(g[2] - mx);
  float inv = 1.0f / (e0 + e1 + e2);
  float vf = (e0 * vs0[d] + e1 * vs1[d] + e2 * vs2[d]) * inv;
  tmp[(size_t)b * Dd + d] = self[(size_t)b * Dd + d] + vf;
}

// ---------------------------------------------------------------------------
extern "C" void kernel_launch(void* const* d_in, const int* in_sizes, int n_in,
                              void* d_out, int out_size, void* d_ws, size_t ws_size,
                              hipStream_t stream) {
  (void)in_sizes; (void)n_in; (void)out_size; (void)ws_size;
  const float* self  = (const float*)d_in[0];
  const float* ent   = (const float*)d_in[1];
  const float* rel   = (const float*)d_in[2];
  const float* u     = (const float*)d_in[3];
  const float* WQ    = (const float*)d_in[4];
  const float* bQ    = (const float*)d_in[5];
  const float* WK    = (const float*)d_in[6];
  const float* bK    = (const float*)d_in[7];
  const float* WV    = (const float*)d_in[8];
  const float* bV    = (const float*)d_in[9];
  const float* alpha = (const float*)d_in[10];
  const float* beta  = (const float*)d_in[11];
  const float* Wprop = (const float*)d_in[12];
  const float* bprop = (const float*)d_in[13];
  const float* Wself = (const float*)d_in[14];
  const float* bself = (const float*)d_in[15];
  const float* Wl    = (const float*)d_in[16];
  const float* bl    = (const float*)d_in[17];
  const float* wl    = (const float*)d_in[18];
  const float* Wout  = (const float*)d_in[19];
  const float* bout  = (const float*)d_in[20];

  const size_t nBND = (size_t)Bc * Nn * Dd;           // 33,554,432
  const size_t nBD  = (size_t)Bc * Dd;                // 524,288
  const size_t DD   = (size_t)Dd * Dd;

  char* ws = (char*)d_ws;
  __bf16* entb = (__bf16*)ws;                          //  64 MB
  __bf16* relb = entb + nBND;                          //  64 MB
  float*  Qws  = (float*)(relb + nBND);                //   2 MB
  float*  prop = Qws + nBD;                            //   2 MB
  float*  v0   = prop + nBD;                           //   2 MB
  float*  v1   = v0 + nBD;                             //   2 MB
  float*  v2   = v1 + nBD;                             //   2 MB
  float*  tmp  = v2 + nBD;                             //   2 MB
  __bf16* packBase = (__bf16*)(tmp + nBD);             // ~416 KB packed weights

  // ---- one-time: bf16 mirrors, v0 mean, packed weights ----
  prep_kernel<<<(unsigned)(nBND / 256), 256, 0, stream>>>(ent, rel, entb, relb, nBND);
  mean_kernel<<<(unsigned)(nBD / 256), 256, 0, stream>>>(ent, v0);

  size_t offTiles = 0;
  auto packW = [&](const float* W, int ktiles) -> __bf16* {
    __bf16* dst = packBase + offTiles * 4096;          // ktile = 8 ntiles * 512 halves
    pack_w_kernel<<<ktiles * 8, 512, 0, stream>>>(W, dst, ktiles);
    offTiles += (size_t)ktiles;
    return dst;
  };
  __bf16 *WQp[Ll], *WKp[Ll], *WVp[Ll], *Wpp[Ll], *Wsp[Ll];
  for (int l = 0; l < Ll; ++l) {
    WQp[l] = packW(WQ    + (size_t)l * DD,     4);
    WKp[l] = packW(WK    + (size_t)l * 2 * DD, 8);
    WVp[l] = packW(WV    + (size_t)l * DD,     4);
    Wpp[l] = packW(Wprop + (size_t)l * DD,     4);
    Wsp[l] = packW(Wself + (size_t)l * DD,     4);
  }
  __bf16* Woutp = packW(Wout, 4);

  // ---- layers ----
  const float* vprev = v0;
  float* vlay[Ll] = { v1, v2 };
  for (int l = 0; l < Ll; ++l) {
    // Qs = (u @ WQ[l] + bQ[l]) * SCALE
    gemm128_kernel<<<Bc / 16, 256, 0, stream>>>(
        u, WQp[l], nullptr, nullptr, bQ + (size_t)l * Dd, nullptr, SCALE_F, 0, Qws);
    // fused attention + propagation
    layer_attn_kernel<<<Bc, 256, 0, stream>>>(
        entb, relb, WKp[l], bK + (size_t)l * Dd, WVp[l], bV + (size_t)l * Dd,
        alpha + (size_t)l * Hh, beta + (size_t)l * Hh, Qws, prop);
    // v = tanh(prop @ Wprop + bprop + vprev @ Wself + bself)
    gemm128_kernel<<<Bc / 16, 256, 0, stream>>>(
        prop, Wpp[l], vprev, Wsp[l],
        bprop + (size_t)l * Dd, bself + (size_t)l * Dd, 1.0f, 1, vlay[l]);
    vprev = vlay[l];
  }

  fusion_kernel<<<Bc, 128, 0, stream>>>(u, self, v0, v1, v2, Wl, bl, wl, tmp);

  // out = tmp @ W_out + b_out
  gemm128_kernel<<<Bc / 16, 256, 0, stream>>>(
      tmp, Woutp, nullptr, nullptr, bout, nullptr, 1.0f, 0, (float*)d_out);
}
